// SEQ_TO_SEQ_MODEL_32865089749662
// MI455X (gfx1250) — matrix operand order repaired and checked by reading, not compiled, not run on hardware
//
#include <hip/hip_runtime.h>
#include <hip/hip_bf16.h>

#define BB 64
#define SS 128
#define HH 512
#define VV 32000
#define TT 32
#define FH 2048   // 4H
#define H2 1024   // 2H

typedef __attribute__((ext_vector_type(16))) _Float16 v16h;
typedef __attribute__((ext_vector_type(8)))  _Float16 v8h;
typedef __attribute__((ext_vector_type(8)))  float    v8f;

__device__ __forceinline__ float sigm(float x) { return 1.0f / (1.0f + expf(-x)); }

// ---------------------------------------------------------------------------
// WMMA GEMM: C[M,N] = A[M,K](f16) @ W[N,K]^T(f16) (+ bias[N]), f32 accumulate.
// Register blocking: each wave owns MT M-tiles x 4 N-tiles (MT*4 v8f accs).
// Inner loop per 32-wide K step: 2*MT A loads + 8 W loads + 4*MT v_wmma.
//   MT=1: latency-optimized (recurrent LSTM GEMMs, max waves in flight)
//   MT=4: throughput-optimized (logits / enc_keys; W streamed from L2 once)
// block = 128 threads (4 waves) -> 256 output columns.
// grid.x = N/256, grid.y = M/(16*MT).
// Fragment addressing (contiguous k order, same for A and B):
//   A (16x32 MxK): lane m=lane%16; elems 0..7  -> K = k0 + 16*(lane>=16) + e
//                                  elems 8..15 -> K = k0 + 8 + 16*(lane>=16) + e
//   B (32x16 KxN): lane n=lane%16; elem e -> K = k0 + 16*(lane>=16) + e
//   C/D (16x16):   VGPR r: row = r + 8*(lane>=16), col = lane%16
// ---------------------------------------------------------------------------
template <int MT>
__global__ void gemm_f16_wmma(const _Float16* __restrict__ A, int lda,
                              const _Float16* __restrict__ W, int ldw,
                              float* __restrict__ C, long ldc,
                              const float* __restrict__ bias, int K) {
    const int wave = threadIdx.x >> 5;
    const int lane = threadIdx.x & 31;
    const int l  = lane & 15;
    const int hi = lane >> 4;
    const int tn0 = (blockIdx.x * 4 + wave) * 4;   // first of 4 N-tiles
    const int tm0 = blockIdx.y * MT;               // first of MT M-tiles

    const _Float16* arow[MT];
#pragma unroll
    for (int m = 0; m < MT; ++m)
        arow[m] = A + (long)((tm0 + m) * 16 + l) * lda + hi * 16;
    const _Float16* wrow[4];
#pragma unroll
    for (int i = 0; i < 4; ++i)
        wrow[i] = W + (long)((tn0 + i) * 16 + l) * ldw + hi * 16;

    v8f acc[MT][4] = {};
#pragma unroll 1
    for (int k = 0; k < K; k += 32) {
        v16h a[MT];
#pragma unroll
        for (int m = 0; m < MT; ++m) {
            v8h a0 = *(const v8h*)(arow[m] + k);
            v8h a1 = *(const v8h*)(arow[m] + k + 8);
#pragma unroll
            for (int e = 0; e < 8; ++e) { a[m][e] = a0[e]; a[m][e + 8] = a1[e]; }
        }
        v16h b[4];
#pragma unroll
        for (int i = 0; i < 4; ++i) b[i] = *(const v16h*)(wrow[i] + k);

#pragma unroll
        for (int m = 0; m < MT; ++m)
#pragma unroll
            for (int i = 0; i < 4; ++i)
                acc[m][i] = __builtin_amdgcn_wmma_f32_16x16x32_f16(
                    false, a[m], false, b[i], (short)0, acc[m][i], false, false);
    }

#pragma unroll
    for (int m = 0; m < MT; ++m)
#pragma unroll
        for (int i = 0; i < 4; ++i) {
            const int col = (tn0 + i) * 16 + l;
            const float bv = bias ? bias[col] : 0.0f;
#pragma unroll
            for (int r = 0; r < 8; ++r) {
                const int row = (tm0 + m) * 16 + r + hi * 8;
                C[(long)row * ldc + col] = acc[m][i][r] + bv;
            }
        }
}

// ---------------------------------------------------------------------------
// Utility / pointwise kernels
// ---------------------------------------------------------------------------
__global__ void cvt_f32_f16(const float* __restrict__ src, _Float16* __restrict__ dst, int n) {
    int i = blockIdx.x * blockDim.x + threadIdx.x;
    if (i < n) dst[i] = (_Float16)src[i];
}

// Wcat[4H, 2H]: first H cols = Wih, last H cols = Whh  (gates = [x|h] @ Wcat^T)
__global__ void build_wcat_enc(const float* __restrict__ Wih, const float* __restrict__ Whh,
                               _Float16* __restrict__ Wcat) {
    int i = blockIdx.x * blockDim.x + threadIdx.x;   // FH*H2 elements
    if (i >= FH * H2) return;
    int n = i >> 10, k = i & (H2 - 1);
    float v = (k < HH) ? Wih[n * HH + k] : Whh[n * HH + (k - HH)];
    Wcat[i] = (_Float16)v;
}

__global__ void zero_f32(float* __restrict__ p, int n) {
    int i = blockIdx.x * blockDim.x + threadIdx.x;
    if (i < n) p[i] = 0.0f;
}

__global__ void zero_f16(_Float16* __restrict__ p, int n) {
    int i = blockIdx.x * blockDim.x + threadIdx.x;
    if (i < n) p[i] = (_Float16)0.0f;
}

__global__ void init_tok(int* __restrict__ tok) {
    if (threadIdx.x < BB) tok[threadIdx.x] = 0;
}

// Astep[b, 0..H) = enc_emb[input_encoder[b, t]]; Astep[b, H..2H) = h[b]   (f16)
__global__ void enc_build_x(const int* __restrict__ ids, int t,
                            const float* __restrict__ emb, const _Float16* __restrict__ h,
                            _Float16* __restrict__ Astep) {
    int i = blockIdx.x * blockDim.x + threadIdx.x;   // B*2H
    if (i >= BB * H2) return;
    int b = i >> 10, j = i & (H2 - 1);
    Astep[i] = (j < HH) ? (_Float16)emb[(long)ids[b * SS + t] * HH + j]
                        : h[b * HH + (j - HH)];
}

// x[b, 0..H) = dec_emb[tok[b]]; x[b, H..2H) = ctx[b]   (f16)
__global__ void dec_build_x(const int* __restrict__ tok, const float* __restrict__ emb,
                            const float* __restrict__ ctx, _Float16* __restrict__ x) {
    int i = blockIdx.x * blockDim.x + threadIdx.x;   // B*2H
    if (i >= BB * H2) return;
    int b = i >> 10, j = i & (H2 - 1);
    x[i] = (_Float16)((j < HH) ? emb[(long)tok[b] * HH + j] : ctx[b * HH + (j - HH)]);
}

// Encoder LSTM pointwise: gates[B,4H]+bias -> update h(f16), c(f32);
// write enc_out32 (f32, attention) and enc_out16 (f16, GEMM A operand).
__global__ void lstm_point_enc(const float* __restrict__ gates, const float* __restrict__ bias,
                               _Float16* __restrict__ h, float* __restrict__ c,
                               float* __restrict__ enc_out32,
                               _Float16* __restrict__ enc_out16, int t) {
    int i = blockIdx.x * blockDim.x + threadIdx.x;   // B*H
    if (i >= BB * HH) return;
    int b = i >> 9, j = i & (HH - 1);
    const float* g = gates + b * FH;
    float gi = g[j]          + bias[j];
    float gf = g[j + HH]     + bias[j + HH];
    float gg = g[j + 2*HH]   + bias[j + 2*HH];
    float go = g[j + 3*HH]   + bias[j + 3*HH];
    float cn = sigm(gf) * c[i] + sigm(gi) * tanhf(gg);
    float hn = sigm(go) * tanhf(cn);
    c[i] = cn;
    h[i] = (_Float16)hn;
    long o = ((long)b * SS + t) * HH + j;
    enc_out32[o] = hn;
    enc_out16[o] = (_Float16)hn;
}

// Decoder LSTM pointwise: h1(f16) = cell(G1 + Hdec + bias, enc_c)
__global__ void lstm_point_dec(const float* __restrict__ G1, const float* __restrict__ Hdec,
                               const float* __restrict__ bias, const float* __restrict__ enc_c,
                               _Float16* __restrict__ h1) {
    int i = blockIdx.x * blockDim.x + threadIdx.x;   // B*H
    if (i >= BB * HH) return;
    int b = i >> 9, j = i & (HH - 1);
    const float* g1 = G1 + b * FH;
    const float* g2 = Hdec + b * FH;
    float gi = g1[j]          + g2[j]          + bias[j];
    float gf = g1[j + HH]     + g2[j + HH]     + bias[j + HH];
    float gg = g1[j + 2*HH]   + g2[j + 2*HH]   + bias[j + 2*HH];
    float go = g1[j + 3*HH]   + g2[j + 3*HH]   + bias[j + 3*HH];
    float cn = sigm(gf) * enc_c[i] + sigm(gi) * tanhf(gg);
    h1[i] = (_Float16)(sigm(go) * tanhf(cn));
}

// Bahdanau attention, one block per batch element.
__global__ void attention_kernel(const float* __restrict__ enc_keys,
                                 const float* __restrict__ enc_out,
                                 const float* __restrict__ q,
                                 const float* __restrict__ att_v,
                                 const float* __restrict__ att_bv,
                                 float* __restrict__ ctx) {
    const int b = blockIdx.x;
    __shared__ float sc[SS];
    __shared__ float qs[HH];
    __shared__ float vs[HH];
    __shared__ float red[2];

    for (int j = threadIdx.x; j < HH; j += blockDim.x) {
        qs[j] = q[b * HH + j];
        vs[j] = att_v[j];
    }
    __syncthreads();

    for (int s = threadIdx.x; s < SS; s += blockDim.x) {
        const float* kr = enc_keys + ((long)b * SS + s) * HH;
        float acc = 0.0f;
        for (int j = 0; j < HH; ++j) acc += vs[j] * tanhf(kr[j] + qs[j]);
        sc[s] = acc + att_bv[0];
    }
    __syncthreads();

    if (threadIdx.x == 0) {
        float m = sc[0];
        for (int s = 1; s < SS; ++s) m = fmaxf(m, sc[s]);
        float sum = 0.0f;
        for (int s = 0; s < SS; ++s) sum += expf(sc[s] - m);
        red[0] = m;
        red[1] = 1.0f / sum;
    }
    __syncthreads();
    const float m = red[0], inv = red[1];
    for (int s = threadIdx.x; s < SS; s += blockDim.x) sc[s] = expf(sc[s] - m) * inv;
    __syncthreads();

    for (int j = threadIdx.x; j < HH; j += blockDim.x) {
        float acc = 0.0f;
        for (int s = 0; s < SS; ++s) acc += sc[s] * enc_out[((long)b * SS + s) * HH + j];
        ctx[b * HH + j] = acc;
    }
}

// argmax over V logits, one block per batch element; logits row = base + b*T*V
__global__ void argmax_kernel(const float* __restrict__ logits, int* __restrict__ tok) {
    const int b = blockIdx.x;
    const float* row = logits + (long)b * TT * VV;
    float best = -3.4e38f;
    int bi = 0;
    for (int v = threadIdx.x; v < VV; v += blockDim.x) {
        float x = row[v];
        if (x > best) { best = x; bi = v; }
    }
    __shared__ float bv[256];
    __shared__ int   bx[256];
    bv[threadIdx.x] = best;
    bx[threadIdx.x] = bi;
    __syncthreads();
    for (int o = 128; o > 0; o >>= 1) {
        if (threadIdx.x < o) {
            float ov = bv[threadIdx.x + o];
            int   oi = bx[threadIdx.x + o];
            if (ov > bv[threadIdx.x] || (ov == bv[threadIdx.x] && oi < bx[threadIdx.x])) {
                bv[threadIdx.x] = ov;
                bx[threadIdx.x] = oi;
            }
        }
        __syncthreads();
    }
    if (threadIdx.x == 0) tok[b] = bx[0];
}

// ---------------------------------------------------------------------------
// Host side
// ---------------------------------------------------------------------------
static inline void launch_gemm_lat(const _Float16* A, int lda, const _Float16* W, int ldw,
                                   float* C, long ldc, const float* bias,
                                   int M, int N, int K, hipStream_t s) {
    dim3 grid(N / 256, M / 16);
    gemm_f16_wmma<1><<<grid, 128, 0, s>>>(A, lda, W, ldw, C, ldc, bias, K);
}

static inline void launch_gemm_tp(const _Float16* A, int lda, const _Float16* W, int ldw,
                                  float* C, long ldc, const float* bias,
                                  int M, int N, int K, hipStream_t s) {
    dim3 grid(N / 256, M / 64);
    gemm_f16_wmma<4><<<grid, 128, 0, s>>>(A, lda, W, ldw, C, ldc, bias, K);
}

extern "C" void kernel_launch(void* const* d_in, const int* in_sizes, int n_in,
                              void* d_out, int out_size, void* d_ws, size_t ws_size,
                              hipStream_t stream) {
    const int*   ids      = (const int*)d_in[0];
    const float* enc_emb  = (const float*)d_in[1];
    const float* dec_emb  = (const float*)d_in[2];
    const float* enc_Wih  = (const float*)d_in[3];
    const float* enc_Whh  = (const float*)d_in[4];
    const float* enc_b    = (const float*)d_in[5];
    const float* dec_Wih  = (const float*)d_in[6];
    const float* dec_Whh  = (const float*)d_in[7];
    const float* dec_b    = (const float*)d_in[8];
    const float* att_W1   = (const float*)d_in[9];
    const float* att_b1   = (const float*)d_in[10];
    const float* att_W2   = (const float*)d_in[11];
    const float* att_b2   = (const float*)d_in[12];
    const float* att_v    = (const float*)d_in[13];
    const float* att_bv   = (const float*)d_in[14];
    const float* out_W    = (const float*)d_in[15];
    const float* out_b    = (const float*)d_in[16];
    float* out = (float*)d_out;

    // ---- carve workspace ----
    char* ws = (char*)d_ws;
    size_t off = 0;
    auto take = [&](size_t bytes) -> char* {
        char* p = ws + off;
        off = (off + bytes + 255) & ~(size_t)255;
        return p;
    };
    _Float16* WcatEnc  = (_Float16*)take((size_t)FH * H2 * 2);   // [4H,2H]
    _Float16* WdecIh   = (_Float16*)take((size_t)FH * H2 * 2);   // [4H,2H]
    _Float16* WdecHh   = (_Float16*)take((size_t)FH * HH * 2);   // [4H,H]
    _Float16* Watt1    = (_Float16*)take((size_t)HH * HH * 2);
    _Float16* Watt2    = (_Float16*)take((size_t)HH * HH * 2);
    _Float16* Wout     = (_Float16*)take((size_t)VV * HH * 2);
    _Float16* h        = (_Float16*)take((size_t)BB * HH * 2);       // f16 state
    _Float16* Astep    = (_Float16*)take((size_t)BB * H2 * 2);
    _Float16* xdec     = (_Float16*)take((size_t)BB * H2 * 2);
    _Float16* h1       = (_Float16*)take((size_t)BB * HH * 2);
    _Float16* enc_out16= (_Float16*)take((size_t)BB * SS * HH * 2);
    float* c        = (float*)take((size_t)BB * HH * 4);
    float* gates    = (float*)take((size_t)BB * FH * 4);
    float* enc_out32= (float*)take((size_t)BB * SS * HH * 4);
    float* enc_keys = (float*)take((size_t)BB * SS * HH * 4);
    float* Hdec     = (float*)take((size_t)BB * FH * 4);
    float* q        = (float*)take((size_t)BB * HH * 4);
    float* ctx      = (float*)take((size_t)BB * HH * 4);
    int*   tok      = (int*)take(256);
    (void)ws_size; (void)n_in; (void)in_sizes; (void)out_size;

    const int TPB = 256;
    auto blocks = [](int n) { return (n + 255) / 256; };

    // ---- weight conversion (f32 -> f16) ----
    build_wcat_enc<<<blocks(FH * H2), TPB, 0, stream>>>(enc_Wih, enc_Whh, WcatEnc);
    cvt_f32_f16<<<blocks(FH * H2), TPB, 0, stream>>>(dec_Wih, WdecIh, FH * H2);
    cvt_f32_f16<<<blocks(FH * HH), TPB, 0, stream>>>(dec_Whh, WdecHh, FH * HH);
    cvt_f32_f16<<<blocks(HH * HH), TPB, 0, stream>>>(att_W1, Watt1, HH * HH);
    cvt_f32_f16<<<blocks(HH * HH), TPB, 0, stream>>>(att_W2, Watt2, HH * HH);
    cvt_f32_f16<<<blocks(VV * HH), TPB, 0, stream>>>(out_W, Wout, VV * HH);

    // ---- init state ----
    zero_f16<<<blocks(BB * HH), TPB, 0, stream>>>(h, BB * HH);
    zero_f32<<<blocks(BB * HH), TPB, 0, stream>>>(c, BB * HH);
    zero_f32<<<blocks(BB * HH), TPB, 0, stream>>>(ctx, BB * HH);
    init_tok<<<1, 64, 0, stream>>>(tok);

    // ---- encoder: 128 sequential LSTM steps (latency-optimized GEMM) ----
    for (int t = 0; t < SS; ++t) {
        enc_build_x<<<blocks(BB * H2), TPB, 0, stream>>>(ids, t, enc_emb, h, Astep);
        // gates[B,4H] = [x_t | h] @ [Wih | Whh]^T
        launch_gemm_lat(Astep, H2, WcatEnc, H2, gates, FH, nullptr, BB, FH, H2, stream);
        lstm_point_enc<<<blocks(BB * HH), TPB, 0, stream>>>(gates, enc_b, h, c,
                                                            enc_out32, enc_out16, t);
    }

    // ---- attention keys: enc_keys[B*S,H] = enc_out @ att_W1^T + att_b1 ----
    launch_gemm_tp(enc_out16, HH, Watt1, HH, enc_keys, HH, att_b1, BB * SS, HH, HH, stream);

    // ---- decoder recurrent-gate precompute (enc_h is the h input EVERY step) ----
    launch_gemm_lat(h, HH, WdecHh, HH, Hdec, FH, nullptr, BB, FH, HH, stream);

    // ---- decoder: 32 autoregressive steps ----
    for (int t = 0; t < TT; ++t) {
        dec_build_x<<<blocks(BB * H2), TPB, 0, stream>>>(tok, dec_emb, ctx, xdec);
        launch_gemm_lat(xdec, H2, WdecIh, H2, gates, FH, nullptr, BB, FH, H2, stream);
        lstm_point_dec<<<blocks(BB * HH), TPB, 0, stream>>>(gates, Hdec, dec_b, c, h1);
        launch_gemm_lat(h1, HH, Watt2, HH, q, HH, att_b2, BB, HH, HH, stream);
        attention_kernel<<<BB, TPB, 0, stream>>>(enc_keys, enc_out32, q, att_v, att_bv, ctx);
        // logits -> d_out[b*T*V + t*V + v]  (throughput GEMM: W_out read once)
        launch_gemm_tp(h1, HH, Wout, HH, out + (long)t * VV, (long)TT * VV, out_b,
                       BB, VV, HH, stream);
        argmax_kernel<<<BB, TPB, 0, stream>>>(out + (long)t * VV, tok);
    }
}
